// ProteinEncoder_70471823393226
// MI455X (gfx1250) — matrix operand order repaired and checked by reading, not compiled, not run on hardware
//
#include <hip/hip_runtime.h>
#include <math.h>

// ---------------- problem constants ----------------
constexpr int NA_  = 8000;
constexpr int NV_  = 4000;
constexpr int N_   = 12000;          // NA_ + NV_
constexpr int FEAT = 32;
constexpr int H_   = 4;
constexpr int D_   = 16;
constexpr int HD_  = 64;
constexpr int LAYERS = 12;           // BLOCKS * DEPTH
constexpr int AK  = 10;
constexpr int AVK = 15;
constexpr int VK  = 15;
constexpr int EAA = NA_ * AK;        // 80000
constexpr int EAV = NA_ * AVK;       // 120000
constexpr int EVV = NV_ * VK;        // 60000
constexpr int ETOT = EAA + EAV + EVV;// 260000
constexpr int CATW = 7 * HD_;        // 448
constexpr float EPS_ = 1e-5f;

#define DEVINL __device__ __forceinline__

typedef __attribute__((ext_vector_type(16))) _Float16 v16h;
typedef __attribute__((ext_vector_type(2)))  _Float16 h2;
typedef __attribute__((ext_vector_type(8)))  float    v8f;

// ---------------- WMMA helper (16x16x32 f16 -> f32) ----------------
DEVINL v8f wmma32(v16h a, v16h b, v8f c) {
  return __builtin_amdgcn_wmma_f32_16x16x32_f16(
      /*neg_a=*/false, a, /*neg_b=*/false, b,
      /*c_mod=*/(short)0, c, /*reuse_a=*/false, /*reuse_b=*/false);
}

// A fragment from f16 row-major source (LDS or global), packed pair reads.
// ISA layout: lane l -> M = l&15; half = l>>4; vgpr v: grp=v>>2, w=v&3;
// K pair = k0 + grp*16 + half*8 + w*2.   (lda in elements, even; base 4B-aligned)
DEVINL v16h frag_a16(const _Float16* A, int lda, int k0, int lane) {
  int m = lane & 15, half = lane >> 4;
  v16h a;
#pragma unroll
  for (int v = 0; v < 8; ++v) {
    int grp = v >> 2, w = v & 3;
    int k = k0 + grp * 16 + half * 8 + w * 2;
    h2 p = *(const h2*)(A + m * lda + k);
    a[2 * v]     = p[0];
    a[2 * v + 1] = p[1];
  }
  return a;
}

// B fragment from pre-packed f16 buffer: per (ntile, kstep, lane) 16 contiguous
// halves (32B, 32B-aligned) in exact fragment order -> single v16h load.
DEVINL v16h frag_b_pk(const _Float16* pk, int KS, int nt, int ks, int lane) {
  return *(const v16h*)(pk + (size_t)(((nt * KS + ks) * 32) + lane) * 16);
}

// ---------------- weight pre-pack: f32 [K,64] -> f16 fragment order ----------
// layout: out[((l*4 + nt)*KS + ks)*512 + lane*16 + elem]
// elem -> v = elem>>1, parity = elem&1 ; k = ks*32 + (v>>2)*16 + half*8 + (v&3)*2 + parity ; n = lane&15
__global__ __launch_bounds__(256)
void pack_b_kernel(const float* __restrict__ W, _Float16* __restrict__ out,
                   int layers, int K) {
  int KS = K / 32;
  int per_layer = 4 * KS * 512;
  int total = layers * per_layer;
  int gid = blockIdx.x * 256 + threadIdx.x;
  if (gid >= total) return;
  int l = gid / per_layer;
  int r = gid % per_layer;
  int elem = r & 15;
  int lane = (r >> 4) & 31;
  int ntks = r >> 9;
  int ks = ntks % KS;
  int nt = ntks / KS;
  int n = lane & 15, half = lane >> 4;
  int v = elem >> 1;
  int k = ks * 32 + (v >> 2) * 16 + half * 8 + (v & 3) * 2 + (elem & 1);
  out[gid] = (_Float16)W[(size_t)l * K * HD_ + (size_t)k * HD_ + nt * 16 + n];
}

// ---------------- input projection: [M,32] @ [32,64] + bias ----------------
__global__ __launch_bounds__(256)
void inproj_kernel(const float* __restrict__ xin, const _Float16* __restrict__ Wpk,
                   const float* __restrict__ bias, float* __restrict__ xout) {
  __shared__ _Float16 t16[32][40];    // stride 40 halves = 20 dwords -> conflict-free
  int tid = threadIdx.x;
  int row0 = blockIdx.x * 32;
  for (int i = tid; i < 32 * 32; i += 256) {
    int rr = i >> 5, cc = i & 31;
    t16[rr][cc] = (_Float16)xin[(row0 + rr) * FEAT + cc];
  }
  __syncthreads();
  int wave = tid >> 5, lane = tid & 31;
  int mt = wave >> 2, nt = wave & 3;
  v8f c = {};
  v16h a = frag_a16(&t16[mt * 16][0], 40, 0, lane);
  v16h b = frag_b_pk(Wpk, 1, nt, 0, lane);
  c = wmma32(a, b, c);
  int n = lane & 15, half = lane >> 4;
#pragma unroll
  for (int r = 0; r < 8; ++r) {
    int m = row0 + mt * 16 + r + 8 * half;
    int col = nt * 16 + n;
    xout[m * HD_ + col] = c[r] + bias[col];
  }
}

// ---------------- kNN: brute force, register top-k ----------------
template <int K>
__global__ __launch_bounds__(256)
void knn_kernel(const float* __restrict__ qpos, int nq,
                const float* __restrict__ cpos, int nc,
                int coff, int* __restrict__ src_out) {
  __shared__ float cpx[256], cpy[256], cpz[256];
  int q = blockIdx.x * 256 + threadIdx.x;
  bool active = q < nq;
  float qx = 0.f, qy = 0.f, qz = 0.f;
  if (active) { qx = qpos[q * 3]; qy = qpos[q * 3 + 1]; qz = qpos[q * 3 + 2]; }
  float bd[K]; int bi[K];
#pragma unroll
  for (int j = 0; j < K; ++j) { bd[j] = 3.4e38f; bi[j] = 0; }
  for (int base = 0; base < nc; base += 256) {
    int nload = nc - base; if (nload > 256) nload = 256;
    __syncthreads();
    if ((int)threadIdx.x < nload) {
      cpx[threadIdx.x] = cpos[(base + threadIdx.x) * 3];
      cpy[threadIdx.x] = cpos[(base + threadIdx.x) * 3 + 1];
      cpz[threadIdx.x] = cpos[(base + threadIdx.x) * 3 + 2];
    }
    __syncthreads();
    if (active) {
      for (int jj = 0; jj < nload; ++jj) {
        float dx = qx - cpx[jj], dy = qy - cpy[jj], dz = qz - cpz[jj];
        float d2 = dx * dx + dy * dy + dz * dz;
        if (d2 < bd[K - 1]) {
          float dcur = d2; int icur = base + jj;
#pragma unroll
          for (int j = 0; j < K; ++j) {
            if (dcur < bd[j]) {
              float tf = bd[j]; bd[j] = dcur; dcur = tf;
              int ti = bi[j]; bi[j] = icur; icur = ti;
            }
          }
        }
      }
    }
  }
  if (active) {
#pragma unroll
    for (int j = 0; j < K; ++j) src_out[q * K + j] = bi[j] + coff;
  }
}

// ---------------- per-edge scalar MLP (1 -> 8 -> 1) ----------------
DEVINL void load_pos(const float* apos, const float* vpos, int i,
                     float& x, float& y, float& z) {
  if (i < NA_) { x = apos[i * 3]; y = apos[i * 3 + 1]; z = apos[i * 3 + 2]; }
  else { int v = i - NA_; x = vpos[v * 3]; y = vpos[v * 3 + 1]; z = vpos[v * 3 + 2]; }
}

__global__ __launch_bounds__(256)
void eattr_kernel(const int* __restrict__ src,
                  const float* __restrict__ apos, const float* __restrict__ vpos,
                  const float* __restrict__ w1, const float* __restrict__ b1,
                  const float* __restrict__ w2, const float* __restrict__ b2,
                  float* __restrict__ eattr) {
  int e = blockIdx.x * 256 + threadIdx.x;
  if (e >= ETOT) return;
  int mi, dst;
  if (e < EAA)            { mi = 0; dst = e / AK; }
  else if (e < EAA + EAV) { mi = 1; dst = (e - EAA) / AVK; }
  else                    { mi = 2; dst = NA_ + (e - EAA - EAV) / VK; }
  int s = src[e];
  float sx, sy, sz, dx_, dy_, dz_;
  load_pos(apos, vpos, s, sx, sy, sz);
  load_pos(apos, vpos, dst, dx_, dy_, dz_);
  float ddx = sx - dx_, ddy = sy - dy_, ddz = sz - dz_;
  float dist = sqrtf(ddx * ddx + ddy * ddy + ddz * ddz);
  float acc = b2[mi];
#pragma unroll
  for (int k = 0; k < 8; ++k) {
    float hh = dist * w1[mi * 8 + k] + b1[mi * 8 + k];
    hh = fmaxf(hh, 0.f);
    acc += hh * w2[mi * 8 + k];
  }
  eattr[e] = acc;
}

// ---------------- LayerNorm + dual GEMM (s = h@Ws, d = h@Wd) ----------------
__global__ __launch_bounds__(256)
void ln_gemm_kernel(const float* __restrict__ xin,
                    const _Float16* __restrict__ Ws_pk, const _Float16* __restrict__ Wd_pk,
                    const float* __restrict__ g, const float* __restrict__ bta,
                    float* __restrict__ s_out, float* __restrict__ d_out_) {
  __shared__ _Float16 h16[32][72];  // stride 72 halves = 36 dwords -> conflict-free
  int tid = threadIdx.x;
  int row0 = blockIdx.x * 32;
  int node = tid >> 3;              // 32 nodes, 8 threads each
  int sub = tid & 7;
  int r = row0 + node;
  float vals[8];
  float sum = 0.f;
#pragma unroll
  for (int j = 0; j < 8; ++j) { vals[j] = xin[r * HD_ + sub * 8 + j]; sum += vals[j]; }
#pragma unroll
  for (int off = 1; off < 8; off <<= 1) sum += __shfl_xor(sum, off, 32);
  float mu = sum * (1.f / 64.f);
  float vs = 0.f;
#pragma unroll
  for (int j = 0; j < 8; ++j) { float t = vals[j] - mu; vs += t * t; }
#pragma unroll
  for (int off = 1; off < 8; off <<= 1) vs += __shfl_xor(vs, off, 32);
  float rstd = rsqrtf(vs * (1.f / 64.f) + EPS_);
#pragma unroll
  for (int j = 0; j < 8; j += 2) {
    int c = sub * 8 + j;
    h2 p;
    p[0] = (_Float16)((vals[j] - mu) * rstd * g[c] + bta[c]);
    p[1] = (_Float16)((vals[j + 1] - mu) * rstd * g[c + 1] + bta[c + 1]);
    *(h2*)&h16[node][c] = p;        // packed ds_store_b32
  }
  __syncthreads();

  int wave = tid >> 5, lane = tid & 31;
  const _Float16* Wpk = (wave < 4) ? Ws_pk : Wd_pk;
  float* out = (wave < 4) ? s_out : d_out_;
  int nt = wave & 3;
#pragma unroll
  for (int mt = 0; mt < 2; ++mt) {
    v8f c = {};
#pragma unroll
    for (int ks = 0; ks < 2; ++ks) {
      v16h a = frag_a16(&h16[mt * 16][0], 72, ks * 32, lane);
      v16h b = frag_b_pk(Wpk, 2, nt, ks, lane);
      c = wmma32(a, b, c);
    }
    int n = lane & 15, half = lane >> 4;
#pragma unroll
    for (int rr = 0; rr < 8; ++rr) {
      int m = row0 + mt * 16 + rr + 8 * half;
      out[m * HD_ + nt * 16 + n] = c[rr];
    }
  }
}

// ------- per-(dst,head) online-softmax edge aggregation + residual relu -------
__global__ __launch_bounds__(256)
void gat_agg_kernel(const float* __restrict__ xin,
                    const float* __restrict__ s, const float* __restrict__ dd,
                    const int* __restrict__ src, const float* __restrict__ eattr,
                    const float* __restrict__ We_l, const float* __restrict__ att_l,
                    float* __restrict__ xout, _Float16* __restrict__ ckpt) {
  int gid = blockIdx.x * 256 + threadIdx.x;
  if (gid >= N_ * H_) return;
  int n = gid >> 2;
  int head = gid & 3;
  int base = head * D_;
  float dv[D_], av[D_], wev[D_], macc[D_];
#pragma unroll
  for (int j = 0; j < D_; ++j) {
    dv[j] = dd[n * HD_ + base + j];
    av[j] = att_l[base + j];
    wev[j] = We_l[base + j];
    macc[j] = 0.f;
  }
  int e0, c0, e1 = 0, c1 = 0;
  if (n < NA_) { e0 = n * AK; c0 = AK; e1 = EAA + n * AVK; c1 = AVK; }
  else         { int v = n - NA_; e0 = EAA + EAV + v * VK; c0 = VK; }
  float mx = -3.4e38f, den = 0.f;
  for (int rg = 0; rg < 2; ++rg) {
    int eb = rg ? e1 : e0;
    int cnt = rg ? c1 : c0;
    for (int t = 0; t < cnt; ++t) {
      int e = eb + t;
      int se = src[e];
      float ea = eattr[e];
      const float* sp = &s[se * HD_ + base];
      float sc = 0.f;
      float sv[D_];
#pragma unroll
      for (int j = 0; j < D_; ++j) {
        float sj = sp[j];
        sv[j] = sj;
        float m = sj + dv[j] + ea * wev[j];
        float lr = (m > 0.f) ? m : 0.2f * m;      // leaky_relu(0.2)
        sc += lr * av[j];
      }
      if (sc > mx) {
        float scale = __expf(mx - sc);            // first edge: exp(-inf)=0
        den = den * scale + 1.f;
#pragma unroll
        for (int j = 0; j < D_; ++j) macc[j] = macc[j] * scale + sv[j];
        mx = sc;
      } else {
        float w = __expf(sc - mx);
        den += w;
#pragma unroll
        for (int j = 0; j < D_; ++j) macc[j] += w * sv[j];
      }
    }
  }
  float inv = 1.f / den;
#pragma unroll
  for (int j = 0; j < D_; ++j) {
    float o = fmaxf(xin[n * HD_ + base + j] + macc[j] * inv, 0.f);
    xout[n * HD_ + base + j] = o;
    if (ckpt != nullptr && n >= NA_)
      ckpt[(size_t)(n - NA_) * CATW + base + j] = (_Float16)o;
  }
}

// ---------------- final MLP: relu(cat@Wo1+bo1)@Wo2+bo2 ----------------
__global__ __launch_bounds__(128)
void out_mlp_kernel(const _Float16* __restrict__ cat16,
                    const _Float16* __restrict__ Wo1_pk, const float* __restrict__ bo1,
                    const _Float16* __restrict__ Wo2_pk, const float* __restrict__ bo2,
                    float* __restrict__ enc) {
  __shared__ _Float16 t16[16][72];
  int row0 = blockIdx.x * 16;
  int wave = threadIdx.x >> 5, lane = threadIdx.x & 31;
  int nt = wave;                      // 4 waves -> 4 column tiles
  v8f c = {};
  const _Float16* Abase = cat16 + (size_t)row0 * CATW;
#pragma unroll 2
  for (int ks = 0; ks < CATW / 32; ++ks) {      // 14 k-steps
    v16h a = frag_a16(Abase, CATW, ks * 32, lane);
    v16h b = frag_b_pk(Wo1_pk, CATW / 32, nt, ks, lane);
    c = wmma32(a, b, c);
  }
  int n = lane & 15, half = lane >> 4;
#pragma unroll
  for (int rr = 0; rr < 8; ++rr) {
    int m = rr + 8 * half;
    t16[m][nt * 16 + n] = (_Float16)fmaxf(c[rr] + bo1[nt * 16 + n], 0.f);
  }
  __syncthreads();
  v8f c2 = {};
#pragma unroll
  for (int ks = 0; ks < 2; ++ks) {
    v16h a = frag_a16(&t16[0][0], 72, ks * 32, lane);
    v16h b = frag_b_pk(Wo2_pk, 2, nt, ks, lane);
    c2 = wmma32(a, b, c2);
  }
#pragma unroll
  for (int rr = 0; rr < 8; ++rr) {
    int m = rr + 8 * half;
    enc[(size_t)(row0 + m) * HD_ + nt * 16 + n] = c2[rr] + bo2[nt * 16 + n];
  }
}

// ---------------- small copies ----------------
__global__ __launch_bounds__(256)
void copy_cat0_kernel(const float* __restrict__ x0, _Float16* __restrict__ cat16) {
  int i = blockIdx.x * 256 + threadIdx.x;
  if (i >= NV_ * HD_) return;
  int v = i >> 6, c = i & 63;
  cat16[(size_t)v * CATW + c] = (_Float16)x0[(size_t)(NA_ + v) * HD_ + c];
}

__global__ __launch_bounds__(256)
void copy_pos_kernel(const float* __restrict__ vpos, float* __restrict__ out) {
  int i = blockIdx.x * 256 + threadIdx.x;
  if (i < NV_ * 3) out[i] = vpos[i];
}

// ---------------- host launcher ----------------
extern "C" void kernel_launch(void* const* d_in, const int* in_sizes, int n_in,
                              void* d_out, int out_size, void* d_ws, size_t ws_size,
                              hipStream_t stream) {
  const float* atom_x    = (const float*)d_in[0];
  const float* atom_pos  = (const float*)d_in[1];
  const float* vox_x     = (const float*)d_in[2];
  const float* vox_pos   = (const float*)d_in[3];
  const float* W_atom_in = (const float*)d_in[4];
  const float* b_atom_in = (const float*)d_in[5];
  const float* W_vox_in  = (const float*)d_in[6];
  const float* b_vox_in  = (const float*)d_in[7];
  const float* emlp_w1   = (const float*)d_in[8];
  const float* emlp_b1   = (const float*)d_in[9];
  const float* emlp_w2   = (const float*)d_in[10];
  const float* emlp_b2   = (const float*)d_in[11];
  const float* ln_g      = (const float*)d_in[12];
  const float* ln_b      = (const float*)d_in[13];
  const float* Ws        = (const float*)d_in[14];
  const float* Wd        = (const float*)d_in[15];
  const float* We        = (const float*)d_in[16];
  const float* att       = (const float*)d_in[17];
  const float* Wo1       = (const float*)d_in[18];
  const float* bo1       = (const float*)d_in[19];
  const float* Wo2       = (const float*)d_in[20];
  const float* bo2       = (const float*)d_in[21];

  // -------- workspace layout --------
  // floats: x0, x1, sbuf, dbuf (N_*64 each), eattr (ETOT)
  // ints:   src (ETOT)
  // f16:    cat16 (NV_*448), Ws_pk (12*4096), Wd_pk (12*4096),
  //         WinA_pk (2048), WinV_pk (2048), Wo1_pk (28672), Wo2_pk (4096)
  float* wsf   = (float*)d_ws;
  float* x0    = wsf;
  float* x1    = x0 + (size_t)N_ * HD_;
  float* sbuf  = x1 + (size_t)N_ * HD_;
  float* dbuf  = sbuf + (size_t)N_ * HD_;
  float* eattr = dbuf + (size_t)N_ * HD_;
  int*   src   = (int*)(eattr + ETOT);
  _Float16* cat16   = (_Float16*)(src + ETOT);           // 32B-aligned offset
  _Float16* Ws_pk   = cat16 + (size_t)NV_ * CATW;
  _Float16* Wd_pk   = Ws_pk + (size_t)LAYERS * 4096;
  _Float16* WinA_pk = Wd_pk + (size_t)LAYERS * 4096;
  _Float16* WinV_pk = WinA_pk + 2048;
  _Float16* Wo1_pk  = WinV_pk + 2048;
  _Float16* Wo2_pk  = Wo1_pk + 28672;

  // 0) pre-pack weights to f16 fragment order
  pack_b_kernel<<<(LAYERS * 4096 + 255) / 256, 256, 0, stream>>>(Ws, Ws_pk, LAYERS, HD_);
  pack_b_kernel<<<(LAYERS * 4096 + 255) / 256, 256, 0, stream>>>(Wd, Wd_pk, LAYERS, HD_);
  pack_b_kernel<<<(2048 + 255) / 256, 256, 0, stream>>>(W_atom_in, WinA_pk, 1, FEAT);
  pack_b_kernel<<<(2048 + 255) / 256, 256, 0, stream>>>(W_vox_in, WinV_pk, 1, FEAT);
  pack_b_kernel<<<(28672 + 255) / 256, 256, 0, stream>>>(Wo1, Wo1_pk, 1, CATW);
  pack_b_kernel<<<(4096 + 255) / 256, 256, 0, stream>>>(Wo2, Wo2_pk, 1, HD_);

  // 1) input projections
  inproj_kernel<<<NA_ / 32, 256, 0, stream>>>(atom_x, WinA_pk, b_atom_in, x0);
  inproj_kernel<<<NV_ / 32, 256, 0, stream>>>(vox_x, WinV_pk, b_vox_in,
                                              x0 + (size_t)NA_ * HD_);
  copy_cat0_kernel<<<(NV_ * HD_ + 255) / 256, 256, 0, stream>>>(x0, cat16);

  // 2) kNN edge construction
  knn_kernel<AK><<<(NA_ + 255) / 256, 256, 0, stream>>>(atom_pos, NA_, atom_pos, NA_, 0, src);
  knn_kernel<AVK><<<(NA_ + 255) / 256, 256, 0, stream>>>(atom_pos, NA_, vox_pos, NV_, NA_, src + EAA);
  knn_kernel<VK><<<(NV_ + 255) / 256, 256, 0, stream>>>(vox_pos, NV_, vox_pos, NV_, NA_, src + EAA + EAV);

  // 3) edge attributes
  eattr_kernel<<<(ETOT + 255) / 256, 256, 0, stream>>>(
      src, atom_pos, vox_pos, emlp_w1, emlp_b1, emlp_w2, emlp_b2, eattr);

  // 4) 12 GAT layers (ping-pong x0/x1); checkpoint vox rows after each block
  float* cur = x0;
  float* nxt = x1;
  for (int l = 0; l < LAYERS; ++l) {
    ln_gemm_kernel<<<N_ / 32, 256, 0, stream>>>(
        cur, Ws_pk + (size_t)l * 4096, Wd_pk + (size_t)l * 4096,
        ln_g + (size_t)l * HD_, ln_b + (size_t)l * HD_, sbuf, dbuf);
    _Float16* ck = ((l & 1) == 1) ? (cat16 + (size_t)(l / 2 + 1) * HD_) : nullptr;
    gat_agg_kernel<<<(N_ * H_ + 255) / 256, 256, 0, stream>>>(
        cur, sbuf, dbuf, src, eattr,
        We + (size_t)l * HD_, att + (size_t)l * HD_, nxt, ck);
    float* tmp = cur; cur = nxt; nxt = tmp;
  }

  // 5) final MLP over vox rows -> enc, then append vox_pos
  float* enc = (float*)d_out;
  out_mlp_kernel<<<NV_ / 16, 128, 0, stream>>>(cat16, Wo1_pk, bo1, Wo2_pk, bo2, enc);
  copy_pos_kernel<<<(NV_ * 3 + 255) / 256, 256, 0, stream>>>(vox_pos, enc + (size_t)NV_ * HD_);
}
